// Speller_85298050498901
// MI455X (gfx1250) — matrix operand order repaired and checked by reading, not compiled, not run on hardware
//
#include <hip/hip_runtime.h>
#include <hip/hip_bf16.h>
#include <math.h>

// ---------------------------------------------------------------------------
// Problem constants (from reference): B=32, T=1024, H=1024, V=512, S=128
// ---------------------------------------------------------------------------
#define B_  32
#define T_  1024
#define H_  1024
#define V_  512
#define S_  128
#define K0_ 2560   /* [one_hot(512) | ctx(1024) | h0(1024)] */
#define K1_ 2048   /* [h0(1024) | h1(1024)]  and  [h1 | ctx] */
#define NG_ 4096   /* 4*H gate width */

typedef __attribute__((ext_vector_type(16))) __bf16 v16bf;
typedef __attribute__((ext_vector_type(8)))  float  v8f;
typedef __attribute__((ext_vector_type(4)))  int    v4i;

// CDNA5 async memory->LDS path (ASYNCcnt-tracked, no VGPR staging), if the
// toolchain declares the gfx1250 builtins; otherwise fall back to
// global_load_b128 + ds_store_b128 through registers.
// Builtin signature (from clang diagnostics): first param is int4 in
// address_space(1) ("__device__"), LDS side is int4 in address_space(3).
#if defined(__has_builtin)
#if __has_builtin(__builtin_amdgcn_global_load_async_to_lds_b128) && \
    __has_builtin(__builtin_amdgcn_s_wait_asynccnt)
#define USE_ASYNC_LDS 1
#endif
#endif

#if defined(USE_ASYNC_LDS)
#define GASYNC_B128(gptr, lptr)                                          \
  __builtin_amdgcn_global_load_async_to_lds_b128(                        \
      (__attribute__((address_space(1))) v4i*)(gptr),                    \
      (__attribute__((address_space(3))) v4i*)(lptr), 0, 0)
#endif

static __device__ inline float bf2f(unsigned int u) {
  union { unsigned int i; float f; } x; x.i = u << 16; return x.f;
}
static __device__ inline unsigned short f2bf(float f) {
  union { float f; unsigned int i; } u; u.f = f;
  unsigned int r = u.i + 0x7FFF + ((u.i >> 16) & 1);
  return (unsigned short)(r >> 16);
}
static __device__ inline float sigmoidf_(float x) { return 1.f / (1.f + __expf(-x)); }

// ---------------------------------------------------------------------------
// GEMM tile staging: copy A(32x64) and W(128x64) bf16 tiles into LDS.
// ---------------------------------------------------------------------------
static __device__ inline void gemm_stage(
    const unsigned short* __restrict__ A, const unsigned short* __restrict__ W,
    unsigned short (*Asb)[72], unsigned short (*Wsb)[72],
    int K, int n0, int k0, int tid)
{
  const int row = tid >> 3, cg = tid & 7;
#if defined(USE_ASYNC_LDS)
  GASYNC_B128(A + (long)row * K + k0 + cg * 8, &Asb[row][cg * 8]);
  #pragma unroll
  for (int i = 0; i < 4; ++i) {
    int q = tid + 256 * i;
    int r2 = q >> 3, c2 = q & 7;
    GASYNC_B128(W + (long)(n0 + r2) * K + k0 + c2 * 8, &Wsb[r2][c2 * 8]);
  }
#else
  *(uint4*)&Asb[row][cg * 8] = *(const uint4*)(A + (long)row * K + k0 + cg * 8);
  #pragma unroll
  for (int i = 0; i < 4; ++i) {
    int q = tid + 256 * i;
    int r2 = q >> 3, c2 = q & 7;
    *(uint4*)&Wsb[r2][c2 * 8] =
        *(const uint4*)(W + (long)(n0 + r2) * K + k0 + c2 * 8);
  }
#endif
}

// ---------------------------------------------------------------------------
// GEMM: C[32,N] = A[32,K](bf16) * W[N,K](bf16)^T + bias_a + bias_b
// Block = 256 threads (8 waves). Block tile: M=32 x N=128. Wave tile: 32x16.
// K in chunks of 64 (two 16x16x32 WMMA K-steps), double-buffered LDS so that
// staging of chunk i+1 overlaps WMMA compute on chunk i (1 barrier per chunk).
// ---------------------------------------------------------------------------
__global__ __launch_bounds__(256) void gemm_bf16_wmma(
    const unsigned short* __restrict__ A,   // [32, K]
    const unsigned short* __restrict__ W,   // [N, K] row-major
    const float* __restrict__ bias_a,       // [N] or null
    const float* __restrict__ bias_b,       // [N] or null
    float* __restrict__ C,                  // [32, N]
    int K, int N)
{
  __shared__ unsigned short As[2][32][72];   // stride 72 keeps b128 alignment
  __shared__ unsigned short Ws[2][128][72];

  const int tid  = threadIdx.x;
  const int lane = tid & 31;
  const int wv   = tid >> 5;                // wave id 0..7
  const int n0   = blockIdx.x * 128;

  v8f acc0 = {};                            // M rows 0..15
  v8f acc1 = {};                            // M rows 16..31

  // prologue: stage chunk 0 into buffer 0
  gemm_stage(A, W, As[0], Ws[0], K, n0, 0, tid);
#if defined(USE_ASYNC_LDS)
  __builtin_amdgcn_s_wait_asynccnt(0);
#endif
  __syncthreads();

  const int nchunks = K >> 6;
  for (int ch = 0; ch < nchunks; ++ch) {
    const int cur = ch & 1, nxt = cur ^ 1;
    if (ch + 1 < nchunks)
      gemm_stage(A, W, As[nxt], Ws[nxt], K, n0, (ch + 1) << 6, tid);

    // ---- compute on current buffer: two K=32 WMMA steps ----
    unsigned short (*Asb)[72] = As[cur];
    unsigned short (*Wsb)[72] = Ws[cur];
    #pragma unroll
    for (int ks = 0; ks < 2; ++ks) {
      const int kb   = ks * 32;
      const int half = lane >> 4;           // 0 | 1
      const int mr   = lane & 15;
      union { unsigned int u[8]; v16bf v; } fa0, fa1, fb;
      #pragma unroll
      for (int vi = 0; vi < 8; ++vi) {
        // 16-bit A layout: VGPR vi holds a K pair; lanes 16-31 offset by 8
        int ka = kb + ((vi < 4) ? (2 * vi + half * 8)
                                : (2 * (vi - 4) + half * 8 + 16));
        fa0.u[vi] = *(const unsigned int*)&Asb[mr][ka];
        fa1.u[vi] = *(const unsigned int*)&Asb[16 + mr][ka];
        // 16-bit B layout (32x16): same K order as A (lanes 16-31 offset by 8)
        int kbm = kb + ((vi < 4) ? (2 * vi + half * 8)
                                 : (2 * (vi - 4) + half * 8 + 16));
        fb.u[vi] = *(const unsigned int*)&Wsb[wv * 16 + mr][kbm];
      }
      acc0 = __builtin_amdgcn_wmma_f32_16x16x32_bf16(
          false, fa0.v, false, fb.v, (short)0, acc0, false, false);
      acc1 = __builtin_amdgcn_wmma_f32_16x16x32_bf16(
          false, fa1.v, false, fb.v, (short)0, acc1, false, false);
    }

#if defined(USE_ASYNC_LDS)
    __builtin_amdgcn_s_wait_asynccnt(0);    // staging of nxt visible in LDS
#endif
    __syncthreads();
  }

  // ---- epilogue: D layout = lane gives N, VGPR r gives M (r / 8+r) ----
  int n = n0 + wv * 16 + (lane & 15);
  float bv = 0.f;
  if (bias_a) bv += bias_a[n];
  if (bias_b) bv += bias_b[n];
  int mbase = (lane < 16) ? 0 : 8;
  #pragma unroll
  for (int r = 0; r < 8; ++r) {
    C[(long)(mbase + r)      * N + n] = acc0[r] + bv;
    C[(long)(16 + mbase + r) * N + n] = acc1[r] + bv;
  }
}

// ---------------------------------------------------------------------------
// LSTM cell elementwise update. gates=[i|f|g|o] (fp32). Writes new h (bf16)
// into two packed activation slots.
// ---------------------------------------------------------------------------
__global__ void lstm_update(const float* __restrict__ gates, float* __restrict__ c,
                            unsigned short* __restrict__ dst0, int s0, int o0,
                            unsigned short* __restrict__ dst1, int s1, int o1)
{
  int idx = blockIdx.x * blockDim.x + threadIdx.x;
  if (idx >= B_ * H_) return;
  int b = idx >> 10, j = idx & 1023;
  const float* g = gates + (long)b * NG_;
  float ig = g[j], fg = g[1024 + j], gg = g[2048 + j], og = g[3072 + j];
  float cn = sigmoidf_(fg) * c[idx] + sigmoidf_(ig) * tanhf(gg);
  float hn = sigmoidf_(og) * tanhf(cn);
  c[idx] = cn;
  unsigned short hb = f2bf(hn);
  dst0[(long)b * s0 + o0 + j] = hb;
  dst1[(long)b * s1 + o1 + j] = hb;
}

// ---------------------------------------------------------------------------
// Attention: one block per batch b (256 thr, 8 waves).
//  energy[t] = dot(h1, L[b,t,:]);  softmax; ctx[h] = sum_t score[t]*L[b,t,h]
// Writes scores (fp32 output) and ctx (bf16) into two activation slots.
// ---------------------------------------------------------------------------
__global__ __launch_bounds__(256) void attention_kernel(
    const unsigned short* __restrict__ Lbf,   // [B,T,H] bf16
    const unsigned short* __restrict__ h1bf,  // Aout base; h1 at [b*2048 + 0..1023]
    unsigned short* __restrict__ A0,          // ctx -> A0[b*2560 + 512 + h]
    unsigned short* __restrict__ Aout,        // ctx -> Aout[b*2048 + 1024 + h]
    float* __restrict__ scores_out)           // [B,T] slice of this step
{
  __shared__ float sh_h1[1024];
  __shared__ float sh_e[1024];
  __shared__ float sh_red[256];
  const int b = blockIdx.x, tid = threadIdx.x;
  const int lane = tid & 31, wv = tid >> 5;

  #pragma unroll
  for (int r = 0; r < 4; ++r) {
    int j = tid + 256 * r;
    sh_h1[j] = bf2f(h1bf[(long)b * K1_ + j]);
  }
  __syncthreads();

  const unsigned short* Lb = Lbf + (long)b * T_ * H_;
  // ---- energies: wave wv owns rows [wv*128, wv*128+128) ----
  for (int row = wv * 128; row < wv * 128 + 128; ++row) {
    const unsigned short* Lr = Lb + (long)row * H_;
    float s = 0.f;
    #pragma unroll
    for (int it = 0; it < 8; ++it) {
      int h = it * 128 + lane * 4;
      uint2 p = *(const uint2*)(Lr + h);
      float4 hv = *(const float4*)&sh_h1[h];
      s += bf2f(p.x & 0xffffu) * hv.x + bf2f(p.x >> 16) * hv.y
         + bf2f(p.y & 0xffffu) * hv.z + bf2f(p.y >> 16) * hv.w;
    }
    for (int off = 16; off; off >>= 1) s += __shfl_down(s, off, 32);
    if (lane == 0) sh_e[row] = s;
  }
  __syncthreads();

  // ---- softmax over T=1024 ----
  float m = -1e30f;
  #pragma unroll
  for (int r = 0; r < 4; ++r) m = fmaxf(m, sh_e[tid + 256 * r]);
  sh_red[tid] = m; __syncthreads();
  for (int s2 = 128; s2; s2 >>= 1) {
    if (tid < s2) sh_red[tid] = fmaxf(sh_red[tid], sh_red[tid + s2]);
    __syncthreads();
  }
  const float M = sh_red[0]; __syncthreads();
  float ls = 0.f;
  #pragma unroll
  for (int r = 0; r < 4; ++r) {
    int j = tid + 256 * r;
    float e = __expf(sh_e[j] - M); sh_e[j] = e; ls += e;
  }
  sh_red[tid] = ls; __syncthreads();
  for (int s2 = 128; s2; s2 >>= 1) {
    if (tid < s2) sh_red[tid] += sh_red[tid + s2];
    __syncthreads();
  }
  const float inv = 1.f / sh_red[0]; __syncthreads();
  #pragma unroll
  for (int r = 0; r < 4; ++r) {
    int j = tid + 256 * r;
    float sc = sh_e[j] * inv; sh_e[j] = sc;
    scores_out[(long)b * T_ + j] = sc;
  }
  __syncthreads();

  // ---- context: thread owns 4 consecutive h ----
  const int hb4 = tid * 4;
  float c0 = 0.f, c1 = 0.f, c2 = 0.f, c3 = 0.f;
  for (int t = 0; t < T_; ++t) {
    float s = sh_e[t];
    uint2 p = *(const uint2*)(Lb + (long)t * H_ + hb4);
    c0 += s * bf2f(p.x & 0xffffu);
    c1 += s * bf2f(p.x >> 16);
    c2 += s * bf2f(p.y & 0xffffu);
    c3 += s * bf2f(p.y >> 16);
  }
  unsigned short o0 = f2bf(c0), o1 = f2bf(c1), o2 = f2bf(c2), o3 = f2bf(c3);
  unsigned short* d0 = A0   + (long)b * K0_ + 512  + hb4;
  unsigned short* d1 = Aout + (long)b * K1_ + 1024 + hb4;
  d0[0] = o0; d0[1] = o1; d0[2] = o2; d0[3] = o3;
  d1[0] = o0; d1[1] = o1; d1[2] = o2; d1[3] = o3;
}

// ---------------------------------------------------------------------------
// Output head: log_softmax over V=512 (bias already added in GEMM),
// write preds, argmax -> next one-hot word (bf16) into A0.
// One block per batch, 256 threads x 2 values.
// ---------------------------------------------------------------------------
__global__ __launch_bounds__(256) void output_kernel(
    const float* __restrict__ logits,   // [B,512]
    float* __restrict__ preds_out,      // [B,V] slice of this step
    unsigned short* __restrict__ A0)
{
  __shared__ float sh_v[256];
  __shared__ int   sh_i[256];
  const int b = blockIdx.x, tid = threadIdx.x;
  float x0 = logits[(long)b * V_ + tid];
  float x1 = logits[(long)b * V_ + 256 + tid];

  float bv; int bi;
  if (x0 >= x1) { bv = x0; bi = tid; } else { bv = x1; bi = 256 + tid; }
  sh_v[tid] = bv; sh_i[tid] = bi;
  __syncthreads();
  for (int s = 128; s; s >>= 1) {
    if (tid < s) {
      float v2 = sh_v[tid + s]; int i2 = sh_i[tid + s];
      if (v2 > sh_v[tid] || (v2 == sh_v[tid] && i2 < sh_i[tid])) {
        sh_v[tid] = v2; sh_i[tid] = i2;
      }
    }
    __syncthreads();
  }
  const float M = sh_v[0]; const int amax = sh_i[0];
  __syncthreads();
  float e0 = __expf(x0 - M), e1 = __expf(x1 - M);
  sh_v[tid] = e0 + e1; __syncthreads();
  for (int s = 128; s; s >>= 1) {
    if (tid < s) sh_v[tid] += sh_v[tid + s];
    __syncthreads();
  }
  const float logZ = M + __logf(sh_v[0]);
  preds_out[(long)b * V_ + tid]       = x0 - logZ;
  preds_out[(long)b * V_ + 256 + tid] = x1 - logZ;
  A0[(long)b * K0_ + tid]       = f2bf(tid == amax ? 1.f : 0.f);
  A0[(long)b * K0_ + 256 + tid] = f2bf((256 + tid) == amax ? 1.f : 0.f);
}

// ---------------------------------------------------------------------------
// Setup: fp32 -> bf16 conversions (concatenated weights along K) and state init
// ---------------------------------------------------------------------------
__global__ void convert_concat_bf16(const float* __restrict__ Asrc, int Ka,
                                    const float* __restrict__ Bsrc, int Kb,
                                    unsigned short* __restrict__ dst, long total)
{
  long idx = (long)blockIdx.x * blockDim.x + threadIdx.x;
  long stride = (long)gridDim.x * blockDim.x;
  const int Kt = Ka + Kb;
  for (; idx < total; idx += stride) {
    long n = idx / Kt; int k = (int)(idx % Kt);
    float v = (k < Ka) ? Asrc[n * Ka + k] : Bsrc[n * Kb + (k - Ka)];
    dst[idx] = f2bf(v);
  }
}

__global__ void convert_bf16(const float* __restrict__ src,
                             unsigned short* __restrict__ dst, long total)
{
  long idx = (long)blockIdx.x * blockDim.x + threadIdx.x;
  long stride = (long)gridDim.x * blockDim.x;
  for (; idx < total; idx += stride) dst[idx] = f2bf(src[idx]);
}

__global__ void init_state(unsigned short* __restrict__ A0,
                           unsigned short* __restrict__ A1,
                           unsigned short* __restrict__ Aout,
                           float* __restrict__ c0, float* __restrict__ c1,
                           const float* __restrict__ listener)
{
  int idx = blockIdx.x * blockDim.x + threadIdx.x;
  if (idx < B_ * K0_) {
    int b = idx / K0_, k = idx % K0_;
    float v = 0.f;
    if (k == 0) v = 1.f;                                   // one_hot(0)
    else if (k >= 512 && k < 1536)
      v = listener[(long)b * T_ * H_ + (k - 512)];         // ctx0 = L[:,0,:]
    A0[idx] = f2bf(v);
  }
  if (idx < B_ * K1_) { A1[idx] = 0; Aout[idx] = 0; }
  if (idx < B_ * H_)  { c0[idx] = 0.f; c1[idx] = 0.f; }
}

// ---------------------------------------------------------------------------
// Host driver
// ---------------------------------------------------------------------------
extern "C" void kernel_launch(void* const* d_in, const int* in_sizes, int n_in,
                              void* d_out, int out_size, void* d_ws, size_t ws_size,
                              hipStream_t stream)
{
  const float* listener = (const float*)d_in[0];
  const float* W_ih0 = (const float*)d_in[1];
  const float* W_hh0 = (const float*)d_in[2];
  const float* b_ih0 = (const float*)d_in[3];
  const float* b_hh0 = (const float*)d_in[4];
  const float* W_ih1 = (const float*)d_in[5];
  const float* W_hh1 = (const float*)d_in[6];
  const float* b_ih1 = (const float*)d_in[7];
  const float* b_hh1 = (const float*)d_in[8];
  const float* W_out = (const float*)d_in[9];
  const float* b_out = (const float*)d_in[10];

  float* preds  = (float*)d_out;                       // [S,B,V]
  float* scores = preds + (size_t)S_ * B_ * V_;        // [S,B,T]

  // workspace carve-out (256B aligned)
  size_t off = 0;
  auto take = [&](size_t nbytes) {
    char* p = (char*)d_ws + off;
    off = (off + nbytes + 255) & ~(size_t)255;
    return (void*)p;
  };
  unsigned short* W0cat = (unsigned short*)take((size_t)NG_ * K0_ * 2); // [4096,2560]
  unsigned short* W1cat = (unsigned short*)take((size_t)NG_ * K1_ * 2); // [4096,2048]
  unsigned short* Woutb = (unsigned short*)take((size_t)V_  * K1_ * 2); // [512,2048]
  unsigned short* Lbf   = (unsigned short*)take((size_t)B_ * T_ * H_ * 2);
  unsigned short* A0    = (unsigned short*)take((size_t)B_ * K0_ * 2);
  unsigned short* A1    = (unsigned short*)take((size_t)B_ * K1_ * 2);
  unsigned short* Aout  = (unsigned short*)take((size_t)B_ * K1_ * 2);
  float* gates0 = (float*)take((size_t)B_ * NG_ * 4);
  float* gates1 = (float*)take((size_t)B_ * NG_ * 4);
  float* c0     = (float*)take((size_t)B_ * H_ * 4);
  float* c1     = (float*)take((size_t)B_ * H_ * 4);
  float* logits = (float*)take((size_t)B_ * V_ * 4);

  // ---- one-time (per call) setup: bf16 packing + state init ----
  convert_concat_bf16<<<2048, 256, 0, stream>>>(W_ih0, V_ + H_, W_hh0, H_,
                                                W0cat, (long)NG_ * K0_);
  convert_concat_bf16<<<2048, 256, 0, stream>>>(W_ih1, H_, W_hh1, H_,
                                                W1cat, (long)NG_ * K1_);
  convert_concat_bf16<<<1024, 256, 0, stream>>>(W_out, 2 * H_, nullptr, 0,
                                                Woutb, (long)V_ * K1_);
  convert_bf16<<<4096, 256, 0, stream>>>(listener, Lbf, (long)B_ * T_ * H_);
  init_state<<<(B_ * K0_ + 255) / 256, 256, 0, stream>>>(A0, A1, Aout, c0, c1, listener);

  // ---- sequential decode: 128 steps ----
  for (int s = 0; s < S_; ++s) {
    // LSTM0: gates0 = [ow|ctx|h0] @ [W_ih0|W_hh0]^T + b_ih0 + b_hh0
    gemm_bf16_wmma<<<NG_ / 128, 256, 0, stream>>>(A0, W0cat, b_ih0, b_hh0,
                                                  gates0, K0_, NG_);
    lstm_update<<<(B_ * H_ + 255) / 256, 256, 0, stream>>>(
        gates0, c0, A1, K1_, 0, A0, K0_, 1536);
    // LSTM1: gates1 = [h0|h1] @ [W_ih1|W_hh1]^T + b_ih1 + b_hh1
    gemm_bf16_wmma<<<NG_ / 128, 256, 0, stream>>>(A1, W1cat, b_ih1, b_hh1,
                                                  gates1, K1_, NG_);
    lstm_update<<<(B_ * H_ + 255) / 256, 256, 0, stream>>>(
        gates1, c1, A1, K1_, 1024, Aout, K1_, 0);
    // attention + softmax + context (writes scores output)
    attention_kernel<<<B_, 256, 0, stream>>>(
        Lbf, Aout, A0, Aout, scores + (size_t)s * B_ * T_);
    // output projection: logits = [h1|ctx] @ W_out^T + b_out
    gemm_bf16_wmma<<<V_ / 128, 256, 0, stream>>>(Aout, Woutb, b_out, nullptr,
                                                 logits, K1_, V_);
    // log_softmax + argmax -> preds + next one-hot word
    output_kernel<<<B_, 256, 0, stream>>>(
        logits, preds + (size_t)s * B_ * V_, A0);
  }
}
